// ColorHistogramLoss_69123203662095
// MI455X (gfx1250) — matrix operand order repaired and checked by reading, not compiled, not run on hardware
//
#include <hip/hip_runtime.h>

// ColorHistogramLoss (EMD between soft histograms) for MI455X / gfx1250.
//
// Strategy: hard-bin pixels into 4096 fine sub-bins (LDS atomics, B128 loads),
// then the soft histogram is a 48x4096 * 4096x64 GEMM against a Gaussian
// kernel matrix, executed with v_wmma_f32_16x16x32_f16 (f16 inputs exact for
// counts<=2048, f32 accumulate). Gaussian entries are generated on the fly
// with v_exp_f32 (TRANS class, co-executes with XDL WMMA). Final
// normalize/cumsum/EMD is a single-wave epilogue.

#define NUM_BINS 64
#define FINE     4096
#define HW       (384 * 384)        // 147456 pixels per channel
#define NCH      24                 // B*C = 8*3
#define TOTCH    48                 // pred channels + target channels
#define CHUNK    4096               // pixels per histogram block
#define CHUNKS_PER_CH (HW / CHUNK)  // 36
#define KBLOCKS  8                  // K-dimension split of the GEMM

typedef __attribute__((ext_vector_type(16))) _Float16 v16h;
typedef __attribute__((ext_vector_type(8)))  float    v8f;

__global__ void zero_ws_kernel(unsigned* __restrict__ p, int n) {
  int i = blockIdx.x * blockDim.x + threadIdx.x;
  if (i < n) p[i] = 0u;
}

__device__ __forceinline__ int bin_of(float x) {
  int s = (int)(x * (float)FINE);
  return s < 0 ? 0 : (s > FINE - 1 ? FINE - 1 : s);
}

// One block per (channel, 4096-pixel chunk): B128 vector loads -> LDS
// histogram atomics -> flush with global atomics. This is the only kernel
// touching the 28 MB inputs; float4 loads keep it bandwidth- not issue-bound.
__global__ void fine_hist_kernel(const float* __restrict__ pred,
                                 const float* __restrict__ targ,
                                 unsigned* __restrict__ hist) {
  __shared__ unsigned lh[FINE];
  const int ch    = blockIdx.x / CHUNKS_PER_CH;   // 0..47
  const int chunk = blockIdx.x % CHUNKS_PER_CH;
  const float* src =
      (ch < NCH ? pred + (size_t)ch * HW : targ + (size_t)(ch - NCH) * HW) +
      (size_t)chunk * CHUNK;
  const float4* src4 = (const float4*)src;        // 16B-aligned: HW,CHUNK %4==0

  for (int i = threadIdx.x; i < FINE; i += blockDim.x) lh[i] = 0u;
  __syncthreads();

  // CHUNK/4 = 1024 float4 per block; 256 threads -> 4 each, issued up front.
  float4 v[4];
#pragma unroll
  for (int j = 0; j < 4; ++j) v[j] = src4[threadIdx.x + j * 256];
#pragma unroll
  for (int j = 0; j < 4; ++j) {
    atomicAdd(&lh[bin_of(v[j].x)], 1u);
    atomicAdd(&lh[bin_of(v[j].y)], 1u);
    atomicAdd(&lh[bin_of(v[j].z)], 1u);
    atomicAdd(&lh[bin_of(v[j].w)], 1u);
  }
  __syncthreads();

  unsigned* dst = hist + (size_t)ch * FINE;
  for (int i = threadIdx.x; i < FINE; i += blockDim.x) {
    unsigned c = lh[i];
    if (c) atomicAdd(&dst[i], c);
  }
}

// soft[ch][b] = sum_s count[ch][s] * exp(-(cs - c_b)^2 / denom)
// 12 waves per block: wave -> (m = M-tile 0..2, n = N-tile 0..3).
// KBLOCKS blocks split the K=4096 reduction; partials merged with f32 atomics.
__global__ void wmma_soft_hist_kernel(const unsigned* __restrict__ hist,
                                      float* __restrict__ soft) {
  const int lane    = threadIdx.x & 31;
  const int wv      = threadIdx.x >> 5;   // 0..11
  const int m       = wv >> 2;            // 0..2  (channel tile)
  const int n       = wv & 3;             // 0..3  (bin tile)
  const int half_id = lane >> 4;          // lane group (K-half selector)
  const int lr      = lane & 15;          // row/col within tile

  // A-fragment slot -> K offset (16-bit A 16x32 layout, CDNA5 ISA)
  int kA[16];
#pragma unroll
  for (int i = 0; i < 16; ++i) {
    int v = i >> 1, e = i & 1;
    int k = (v < 4) ? (v * 2 + e) : (16 + (v - 4) * 2 + e);
    kA[i] = k + half_id * 8;
  }

  const float inv_fine = 1.0f / (float)FINE;
  const float bw       = 1.0f / (float)NUM_BINS;
  const float denom    = 2.0f * bw * bw + 1e-7f;
  const float nid      = -1.0f / denom;
  const float cb       = (float)(n * 16 + lr) * (1.0f / 63.0f);  // linspace(0,1,64)

  const unsigned* arow = hist + (size_t)(m * 16 + lr) * FINE;

  const int kc_begin = blockIdx.x * (FINE / 32 / KBLOCKS);   // 16 chunks each
  const int kc_end   = kc_begin + (FINE / 32 / KBLOCKS);

  v8f c = {};
  for (int kc = kc_begin; kc < kc_end; ++kc) {
    const int kbase = kc * 32;
    v16h a, b;
#pragma unroll
    for (int i = 0; i < 16; ++i) {
      // A: counts (exact in f16 up to 2048; actual max ~hundreds)
      a[i] = (_Float16)(float)arow[kbase + kA[i]];
      // B: Gaussian kernel entry, generated on the fly (v_exp_f32)
      float cs = ((float)(kbase + kA[i]) + 0.5f) * inv_fine;
      float d  = cs - cb;
      b[i] = (_Float16)__expf(d * d * nid);
    }
    c = __builtin_amdgcn_wmma_f32_16x16x32_f16(
        /*neg_a=*/false, a, /*neg_b=*/false, b,
        /*c_mod=*/(short)0, c, /*reuse_a=*/false, /*reuse_b=*/false);
  }

#pragma unroll
  for (int r = 0; r < 8; ++r) {
    int ch  = m * 16 + r + 8 * half_id;   // D layout: VGPR r -> M=r / M=8+r
    int bin = n * 16 + lr;
    atomicAdd(&soft[ch * NUM_BINS + bin], c[r]);
  }
}

// One wave: per-channel normalize + cumsum + |dCDF|, then mean over 24*64.
__global__ void emd_kernel(const float* __restrict__ soft,
                           float* __restrict__ out) {
  __shared__ float red[32];
  const int t = threadIdx.x;  // 0..31
  float acc = 0.0f;
  if (t < NCH) {
    const float* p = soft + t * NUM_BINS;
    const float* q = soft + (NCH + t) * NUM_BINS;
    float ps = 0.0f, qs = 0.0f;
    for (int b = 0; b < NUM_BINS; ++b) { ps += p[b]; qs += q[b]; }
    float pinv = 1.0f / (ps + 1e-7f);
    float qinv = 1.0f / (qs + 1e-7f);
    float pc = 0.0f, qc = 0.0f;
    for (int b = 0; b < NUM_BINS; ++b) {
      pc += p[b] * pinv;
      qc += q[b] * qinv;
      acc += fabsf(pc - qc);
    }
  }
  red[t] = acc;
  __syncthreads();
  if (t == 0) {
    float s = 0.0f;
    for (int i = 0; i < 32; ++i) s += red[i];
    out[0] = s / (float)(NCH * NUM_BINS);
  }
}

extern "C" void kernel_launch(void* const* d_in, const int* in_sizes, int n_in,
                              void* d_out, int out_size, void* d_ws, size_t ws_size,
                              hipStream_t stream) {
  (void)in_sizes; (void)n_in; (void)out_size; (void)ws_size;
  const float* pred = (const float*)d_in[0];
  const float* targ = (const float*)d_in[1];

  // Workspace layout: [48*4096 u32 fine hist][48*64 f32 soft hist] ~ 0.8 MB
  unsigned* hist = (unsigned*)d_ws;
  float*    soft = (float*)((char*)d_ws + (size_t)TOTCH * FINE * sizeof(unsigned));

  // Zero both regions every launch (atomic accumulation targets).
  const int nz = TOTCH * FINE + TOTCH * NUM_BINS;  // zero-bits == 0.0f for floats
  zero_ws_kernel<<<(nz + 255) / 256, 256, 0, stream>>>(hist, nz);

  fine_hist_kernel<<<TOTCH * CHUNKS_PER_CH, 256, 0, stream>>>(pred, targ, hist);

  wmma_soft_hist_kernel<<<KBLOCKS, 384, 0, stream>>>(hist, soft);

  emd_kernel<<<1, 32, 0, stream>>>(soft, (float*)d_out);
}
